// MHDM__68968584839218
// MI455X (gfx1250) — matrix operand order repaired and checked by reading, not compiled, not run on hardware
//
#include <hip/hip_runtime.h>
#include <cstdint>

// MI455X / gfx1250, wave32. bf16 WMMA GEMM pipeline for multi-head distance
// attention, with double-buffered LDS and async global->LDS tile copies.

typedef __attribute__((ext_vector_type(4)))  __bf16 v4bf;
typedef __attribute__((ext_vector_type(8)))  __bf16 v8bf;
typedef __attribute__((ext_vector_type(16))) __bf16 v16bf;
typedef __attribute__((ext_vector_type(8)))  float  v8f;

// ---- CDNA5 async global->LDS copy (ASYNCcnt-tracked) ----------------------
// LDS operand is the 32-bit LDS byte offset = low 32 bits of the flat pointer
// (ISA 10.2, LDS aperture).
__device__ __forceinline__ void async_cp_b64(void* lds, const void* gaddr) {
    uint32_t l = (uint32_t)(uintptr_t)lds;
    asm volatile("global_load_async_to_lds_b64 %0, %1, off"
                 :: "v"(l), "v"(gaddr) : "memory");
}
__device__ __forceinline__ void wait_asynccnt0() {
    asm volatile("s_wait_asynccnt 0x0" ::: "memory");
}

// ---------------------------------------------------------------------------
// f32 -> bf16 cast (vectorized float4 -> 4x bf16)
// ---------------------------------------------------------------------------
__global__ __launch_bounds__(256) void cast_f32_bf16(const float* __restrict__ in,
                                                     __bf16* __restrict__ out,
                                                     long long n4) {
    long long i = (long long)blockIdx.x * 256 + threadIdx.x;
    if (i < n4) {
        float4 f = ((const float4*)in)[i];
        v4bf o;
        o[0] = (__bf16)f.x; o[1] = (__bf16)f.y; o[2] = (__bf16)f.z; o[3] = (__bf16)f.w;
        ((v4bf*)out)[i] = o;
    }
}

// ---------------------------------------------------------------------------
// q2[b,h,i] = sum_k qh[i,k]^2   (qh read as bf16, accumulated f32)
// ---------------------------------------------------------------------------
__global__ __launch_bounds__(256) void q2_kernel(const __bf16* __restrict__ qk,
                                                 float* __restrict__ q2) {
    long long tid = (long long)blockIdx.x * 256 + threadIdx.x;   // 0 .. 98303
    int i = (int)(tid % 1536);
    int z = (int)(tid / 1536);
    int h = z & 7, b = z >> 3;
    const __bf16* p = qk + (long long)b * 1572864 + (long long)i * 1024 + h * 128;
    float s = 0.f;
    #pragma unroll 4
    for (int k = 0; k < 128; k += 4) {
        v4bf r = *(const v4bf*)(p + k);
        #pragma unroll
        for (int j = 0; j < 4; ++j) { float v = (float)r[j]; s += v * v; }
    }
    q2[tid] = s;
}

// ---------------------------------------------------------------------------
// Row softmax: 1536 elements per row, one 256-thread block per row.
// ---------------------------------------------------------------------------
__global__ __launch_bounds__(256) void softmax_rows(const float* __restrict__ logits,
                                                    float* __restrict__ attn) {
    long long row = blockIdx.x;
    const float* in = logits + row * 1536;
    float* out = attn + row * 1536;
    __shared__ float red[256];
    int t = threadIdx.x;
    float vals[6];
    float m = -3.4e38f;
    #pragma unroll
    for (int j = 0; j < 6; ++j) { vals[j] = in[t + j * 256]; m = fmaxf(m, vals[j]); }
    red[t] = m; __syncthreads();
    for (int s = 128; s > 0; s >>= 1) {
        if (t < s) red[t] = fmaxf(red[t], red[t + s]);
        __syncthreads();
    }
    m = red[0]; __syncthreads();
    float sum = 0.f;
    #pragma unroll
    for (int j = 0; j < 6; ++j) { vals[j] = __expf(vals[j] - m); sum += vals[j]; }
    red[t] = sum; __syncthreads();
    for (int s = 128; s > 0; s >>= 1) {
        if (t < s) red[t] += red[t + s];
        __syncthreads();
    }
    float inv = 1.0f / red[0];
    #pragma unroll
    for (int j = 0; j < 6; ++j) out[t + j * 256] = vals[j] * inv;
}

// ---------------------------------------------------------------------------
// Tiled bf16 WMMA GEMM, 128x128 C tile, BK=64, double-buffered LDS.
//   A_KM : A element (m,k) at A[k*lda + m]     -> register 4x4-transpose stage
//   A_F32: A is f32 at A[m*lda + k], converted -> register stage
//   else : A bf16 row-major                    -> ASYNC global->LDS copy
//   B_NM : B element (k,n) at B[n*ldb + k]     -> ASYNC global->LDS copy
//   else : B bf16 K-major                      -> register 4x4-transpose stage
//   C_BF16 / BIAS / SCALE2 applied at store: C = (SCALE2?2:1)*acc [- q2r - q2c]
// 8 wave32 waves (4Mx2N), each owns 32x64 = 2x4 WMMA 16x16x32 tiles.
// ---------------------------------------------------------------------------
template<bool A_KM, bool A_F32, bool B_NM, bool C_BF16, bool BIAS, bool SCALE2>
__global__ __launch_bounds__(256) void gemm_wmma_bf16(
    const void* __restrict__ Ag, const __bf16* __restrict__ Bg, void* __restrict__ Cg,
    int K, int lda, int ldb, int ldc,
    long long sa_b, long long sa_h, long long sb_b, long long sb_h,
    long long sc_b, long long sc_h, int zdiv,
    const float* __restrict__ bias, long long sbias)
{
    constexpr bool ASYNC_A = (!A_KM && !A_F32);
    constexpr bool ASYNC_B = B_NM;
    constexpr bool ANY_ASYNC = ASYNC_A || ASYNC_B;

    __shared__ alignas(64) __bf16 As[2][128][80];   // [buf][m][k], 32B-aligned rows
    __shared__ alignas(64) __bf16 Bs[2][128][80];   // [buf][n][k]

    const int t    = threadIdx.x;
    const int lane = t & 31;
    const int wid  = t >> 5;
    const int wm   = wid & 3;        // 4 row-blocks of 32
    const int wn   = wid >> 2;       // 2 col-blocks of 64
    const int hf   = lane >> 4;      // lane half
    const int l16  = lane & 15;

    const int z  = blockIdx.z;
    const int bb = z / zdiv;
    const int hh = z - bb * zdiv;

    const __bf16* Ab = (const __bf16*)Ag + (A_F32 ? 0 : (bb * sa_b + hh * sa_h));
    const float*  Af = (const float*) Ag + (A_F32 ? (bb * sa_b + hh * sa_h) : 0);
    const __bf16* Bp = Bg + bb * sb_b + hh * sb_h;
    __bf16* Cb = (__bf16*)Cg + (C_BF16 ? (bb * sc_b + hh * sc_h) : 0);
    float*  Cf = (float*) Cg + (C_BF16 ? 0 : (bb * sc_b + hh * sc_h));
    const float* bp = BIAS ? (bias + (long long)z * sbias) : nullptr;

    const int m0 = blockIdx.x * 128;
    const int n0 = blockIdx.y * 128;

    v4bf ra[8], rb[8];   // register staging (only used on non-async paths)

    // ---- staging helpers -------------------------------------------------
    auto stageA_load = [&](int kk0) {
        if (A_KM) {                     // K-major: 4x4 blocks, transpose in regs
            #pragma unroll
            for (int i = 0; i < 2; ++i) {
                int blk = t + i * 256;
                int kk = (blk >> 5) << 2;       // 0..60
                int mm = (blk & 31) << 2;       // 0..124
                #pragma unroll
                for (int j = 0; j < 4; ++j)
                    ra[i * 4 + j] = *(const v4bf*)(Ab + (long long)(kk0 + kk + j) * lda + (m0 + mm));
            }
        } else if (A_F32) {             // f32 row-major: convert
            #pragma unroll
            for (int i = 0; i < 8; ++i) {
                int idx = t + i * 256;
                int row = idx >> 4;
                int kc  = (idx & 15) << 2;
                float4 f = *(const float4*)(Af + (long long)(m0 + row) * lda + (kk0 + kc));
                v4bf d;
                d[0] = (__bf16)f.x; d[1] = (__bf16)f.y; d[2] = (__bf16)f.z; d[3] = (__bf16)f.w;
                ra[i] = d;
            }
        }
    };
    auto stageA_store = [&](int bsel) {
        if (A_KM) {
            #pragma unroll
            for (int i = 0; i < 2; ++i) {
                int blk = t + i * 256;
                int kk = (blk >> 5) << 2;
                int mm = (blk & 31) << 2;
                #pragma unroll
                for (int ii = 0; ii < 4; ++ii) {
                    v4bf d;
                    #pragma unroll
                    for (int j = 0; j < 4; ++j) d[j] = ra[i * 4 + j][ii];
                    *(v4bf*)&As[bsel][mm + ii][kk] = d;
                }
            }
        } else if (A_F32) {
            #pragma unroll
            for (int i = 0; i < 8; ++i) {
                int idx = t + i * 256;
                int row = idx >> 4;
                int kc  = (idx & 15) << 2;
                *(v4bf*)&As[bsel][row][kc] = ra[i];
            }
        }
    };
    auto stageA_async = [&](int kk0, int bsel) {
        if (ASYNC_A) {
            #pragma unroll
            for (int i = 0; i < 8; ++i) {
                int idx = t + i * 256;
                int row = idx >> 4;
                int kc  = (idx & 15) << 2;
                async_cp_b64(&As[bsel][row][kc],
                             Ab + (long long)(m0 + row) * lda + (kk0 + kc));
            }
        }
    };
    auto stageB_load = [&](int kk0) {
        if (!B_NM) {                    // K-major: 4x4 blocks, transpose in regs
            #pragma unroll
            for (int i = 0; i < 2; ++i) {
                int blk = t + i * 256;
                int kk = (blk >> 5) << 2;
                int nn = (blk & 31) << 2;
                #pragma unroll
                for (int j = 0; j < 4; ++j)
                    rb[i * 4 + j] = *(const v4bf*)(Bp + (long long)(kk0 + kk + j) * ldb + (n0 + nn));
            }
        }
    };
    auto stageB_store = [&](int bsel) {
        if (!B_NM) {
            #pragma unroll
            for (int i = 0; i < 2; ++i) {
                int blk = t + i * 256;
                int kk = (blk >> 5) << 2;
                int nn = (blk & 31) << 2;
                #pragma unroll
                for (int ii = 0; ii < 4; ++ii) {
                    v4bf d;
                    #pragma unroll
                    for (int j = 0; j < 4; ++j) d[j] = rb[i * 4 + j][ii];
                    *(v4bf*)&Bs[bsel][nn + ii][kk] = d;
                }
            }
        }
    };
    auto stageB_async = [&](int kk0, int bsel) {
        if (ASYNC_B) {
            #pragma unroll
            for (int i = 0; i < 8; ++i) {
                int idx = t + i * 256;
                int row = idx >> 4;
                int kc  = (idx & 15) << 2;
                async_cp_b64(&Bs[bsel][row][kc],
                             Bp + (long long)(n0 + row) * ldb + (kk0 + kc));
            }
        }
    };

    // ---- accumulators ----------------------------------------------------
    v8f acc[2][4];
    #pragma unroll
    for (int tm = 0; tm < 2; ++tm)
        #pragma unroll
        for (int tn = 0; tn < 4; ++tn)
            #pragma unroll
            for (int r = 0; r < 8; ++r) acc[tm][tn][r] = 0.0f;

    // ---- prologue: stage tile 0 into buffer 0 ----------------------------
    stageA_load(0); stageB_load(0);
    stageA_store(0); stageB_store(0);
    stageA_async(0, 0); stageB_async(0, 0);
    if (ANY_ASYNC) wait_asynccnt0();
    __syncthreads();

    int buf = 0;
    for (int k0 = 0; k0 < K; k0 += 64) {
        const bool more = (k0 + 64 < K);
        if (more) {
            // kick off tile k+1 while computing tile k
            stageA_async(k0 + 64, buf ^ 1);
            stageB_async(k0 + 64, buf ^ 1);
            stageA_load(k0 + 64);
            stageB_load(k0 + 64);
        }
        if (k0 + 128 < K) {   // L2 prefetch, distance 2 (global_prefetch_b8)
            int kn = k0 + 128;
            int row = t >> 4, kc = (t & 15) << 2;
            if (A_F32)      __builtin_prefetch(Af + (long long)(m0 + row) * lda + (kn + kc), 0, 1);
            else if (!A_KM) __builtin_prefetch(Ab + (long long)(m0 + row) * lda + (kn + kc), 0, 1);
            else            __builtin_prefetch(Ab + (long long)(kn + (t >> 5)) * lda + (m0 + ((t & 31) << 2)), 0, 1);
            if (B_NM)       __builtin_prefetch(Bp + (long long)(n0 + row) * ldb + (kn + kc), 0, 1);
            else            __builtin_prefetch(Bp + (long long)(kn + (t >> 5)) * ldb + (n0 + ((t & 31) << 2)), 0, 1);
        }

        // ---- compute from LDS[buf]: 2 k-steps of 32, 8 WMMA tiles each ----
        #pragma unroll
        for (int s = 0; s < 2; ++s) {
            v16bf af[2], bfb[4];
            #pragma unroll
            for (int tm = 0; tm < 2; ++tm) {
                // A 16x32 bf16 frag: lane holds M=l16; K = {0..7,16..23} + 8*half
                const __bf16* p = &As[buf][wm * 32 + tm * 16 + l16][s * 32 + hf * 8];
                v8bf lo = *(const v8bf*)p;
                v8bf hi = *(const v8bf*)(p + 16);
                af[tm] = __builtin_shufflevector(lo, hi,
                         0, 1, 2, 3, 4, 5, 6, 7, 8, 9, 10, 11, 12, 13, 14, 15);
            }
            #pragma unroll
            for (int tn = 0; tn < 4; ++tn) {
                const __bf16* q = &Bs[buf][wn * 64 + tn * 16 + l16][s * 32 + hf * 8];
                v8bf blo = *(const v8bf*)q;
                v8bf bhi = *(const v8bf*)(q + 16);
                bfb[tn] = __builtin_shufflevector(blo, bhi,
                          0, 1, 2, 3, 4, 5, 6, 7, 8, 9, 10, 11, 12, 13, 14, 15);
            }
            #pragma unroll
            for (int tm = 0; tm < 2; ++tm)
                #pragma unroll
                for (int tn = 0; tn < 4; ++tn)
                    acc[tm][tn] = __builtin_amdgcn_wmma_f32_16x16x32_bf16(
                        false, af[tm], false, bfb[tn],
                        (short)0, acc[tm][tn], false, false);
        }

        if (!more) break;
        // commit register-staged tile k+1 into the other buffer
        stageA_store(buf ^ 1);
        stageB_store(buf ^ 1);
        if (ANY_ASYNC) wait_asynccnt0();
        __syncthreads();
        buf ^= 1;
    }

    // ---- store C: VGPR r holds rows (hf*8 + r), lanes l16 are columns ----
    #pragma unroll
    for (int tm = 0; tm < 2; ++tm) {
        #pragma unroll
        for (int tn = 0; tn < 4; ++tn) {
            int col = n0 + wn * 64 + tn * 16 + l16;
            #pragma unroll
            for (int r = 0; r < 8; ++r) {
                int row = m0 + wm * 32 + tm * 16 + hf * 8 + r;
                float v = acc[tm][tn][r];
                if (SCALE2) v = 2.0f * v;
                if (BIAS)   v = v - bp[row] - bp[col];
                if (C_BF16) Cb[(long long)row * ldc + col] = (__bf16)v;
                else        Cf[(long long)row * ldc + col] = v;
            }
        }
    }
}

// ---------------------------------------------------------------------------
// Host launcher. B=8, N=1024, D=1536, H=8, Dh=128.
// Outputs: y (B*N*D) | attn (B*H*D*D) | logits (B*H*D*D), all f32.
// ---------------------------------------------------------------------------
extern "C" void kernel_launch(void* const* d_in, const int* in_sizes, int n_in,
                              void* d_out, int out_size, void* d_ws, size_t ws_size,
                              hipStream_t stream) {
    (void)in_sizes; (void)n_in; (void)out_size; (void)ws_size;

    const float* x    = (const float*)d_in[0];
    const float* Wqk  = (const float*)d_in[1];
    const float* Wv   = (const float*)d_in[2];
    const float* Wout = (const float*)d_in[3];

    char* ws = (char*)d_ws;
    __bf16* x_bf    = (__bf16*)(ws);                  // 25165824 B
    __bf16* qk_bf   = (__bf16*)(ws + 25165824LL);     // 25165824 B
    __bf16* v_bf    = (__bf16*)(ws + 50331648LL);     // 25165824 B
    __bf16* w_bf    = (__bf16*)(ws + 75497472LL);     // 25165824 B
    __bf16* wqk_bf  = (__bf16*)(ws + 100663296LL);    // 2097152 B
    __bf16* wv_bf   = (__bf16*)(ws + 102760448LL);    // 2097152 B
    __bf16* wout_bf = (__bf16*)(ws + 104857600LL);    // 2097152 B
    float*  q2      = (float*) (ws + 106954752LL);    // 393216 B

    float* y_out  = (float*)d_out;                       // 12582912
    float* attn   = (float*)d_out + 12582912LL;          // 150994944
    float* logits = (float*)d_out + 163577856LL;         // 150994944

    // 1) casts to bf16
    cast_f32_bf16<<<12288, 256, 0, stream>>>(x,    x_bf,    3145728LL);
    cast_f32_bf16<<<1024,  256, 0, stream>>>(Wqk,  wqk_bf,  262144LL);
    cast_f32_bf16<<<1024,  256, 0, stream>>>(Wv,   wv_bf,   262144LL);
    cast_f32_bf16<<<1024,  256, 0, stream>>>(Wout, wout_bf, 262144LL);

    // 2) qk = x^T @ Wqk, v = x^T @ Wv  (per batch; M=1536, N=1024, K=1024)
    {
        dim3 g(12, 8, 8);
        gemm_wmma_bf16<true, false, false, true, false, false><<<g, 256, 0, stream>>>(
            x_bf, wqk_bf, qk_bf, 1024, 1536, 1024, 1024,
            1572864LL, 0LL, 0LL, 0LL, 1572864LL, 0LL, 1, nullptr, 0LL);
        gemm_wmma_bf16<true, false, false, true, false, false><<<g, 256, 0, stream>>>(
            x_bf, wv_bf, v_bf, 1024, 1536, 1024, 1024,
            1572864LL, 0LL, 0LL, 0LL, 1572864LL, 0LL, 1, nullptr, 0LL);
    }

    // 3) q2 row norms from bf16 qh (consistent with WMMA dot -> exact 0 diagonal)
    q2_kernel<<<384, 256, 0, stream>>>(qk_bf, q2);

    // 4) logits = 2*qh qh^T - q2_i - q2_j  (per b,h; M=N=1536, K=128)
    //    A and B both async global->LDS; x2 applied at store.
    {
        dim3 g(12, 12, 64);
        gemm_wmma_bf16<false, false, true, false, true, true><<<g, 256, 0, stream>>>(
            qk_bf, qk_bf, logits, 128, 1024, 1024, 1536,
            1572864LL, 128LL, 1572864LL, 128LL,
            18874368LL, 2359296LL, 8, q2, 1536LL);
    }

    // 5) attn = softmax(logits) row-wise (B*H*D = 98304 rows of 1536)
    softmax_rows<<<98304, 256, 0, stream>>>(logits, attn);

    // 6) w = attn @ vh  (per b,h; M=1536, N=128, K=1536; attn f32 -> bf16 inline)
    {
        dim3 g(12, 1, 64);
        gemm_wmma_bf16<false, true, false, true, false, false><<<g, 256, 0, stream>>>(
            attn, v_bf, w_bf, 1536, 1536, 1024, 1024,
            18874368LL, 2359296LL, 1572864LL, 128LL,
            1572864LL, 128LL, 8, nullptr, 0LL);
    }

    // 7) y[b][n][d] = sum_m Wout[m,n] * w[b][d][m]  (Wout^T @ w^T), written in
    //    transposed (B,N,D) layout. M=1024, N'=1536, K=1024. B async.
    {
        dim3 g(8, 12, 8);
        gemm_wmma_bf16<true, false, true, false, false, false><<<g, 256, 0, stream>>>(
            wout_bf, w_bf, y_out, 1024, 1024, 1024, 1536,
            0LL, 0LL, 1572864LL, 0LL, 1572864LL, 0LL, 1, nullptr, 0LL);
    }
}
